// SpatialGNN_12867722018827
// MI455X (gfx1250) — matrix operand order repaired and checked by reading, not compiled, not run on hardware
//
#include <hip/hip_runtime.h>
#include <cstddef>

// ---------------------------------------------------------------------------
// Types for CDNA5 WMMA (wave32, gfx1250)
// ---------------------------------------------------------------------------
typedef __attribute__((ext_vector_type(16))) __bf16 v16bf;
typedef __attribute__((ext_vector_type(8)))  float  v8f;

#define HID 64
#define W1O 128           // edge-MLP hidden width (2*H)
#define W2R 4096          // H*H
#define EDGE_DIM 4

__device__ __forceinline__ float sigmoidf_(float x) { return 1.0f / (1.0f + expf(-x)); }
__device__ __forceinline__ float siluf_(float x)    { return x / (1.0f + expf(-x)); }

// ---------------------------------------------------------------------------
// zero fill
// ---------------------------------------------------------------------------
__global__ void zero_f32_kernel(float* __restrict__ p, size_t n) {
    size_t t = (size_t)blockIdx.x * blockDim.x + threadIdx.x;
    if (t < n) p[t] = 0.0f;
}

// ---------------------------------------------------------------------------
// convert nn_W2 (4096x128 f32, row-major) -> bf16
// ---------------------------------------------------------------------------
__global__ void conv_w2_kernel(const float* __restrict__ w2, __bf16* __restrict__ w2b, int n) {
    int t = blockIdx.x * blockDim.x + threadIdx.x;
    if (t < n) w2b[t] = (__bf16)w2[t];
}

// ---------------------------------------------------------------------------
// x = silu(x_in @ fl_W.T + fl_b) ; h = x ; also bf16 mirror
// ---------------------------------------------------------------------------
__global__ void node_embed_kernel(const float* __restrict__ xin, const float* __restrict__ flW,
                                  const float* __restrict__ flb, float* __restrict__ xf,
                                  float* __restrict__ hf, __bf16* __restrict__ xb, int N) {
    int t = blockIdx.x * blockDim.x + threadIdx.x;
    if (t >= N * HID) return;
    int n = t >> 6, c = t & 63;
    float s = flb[c];
    const float* xr = xin + (size_t)n * 11;
    const float* wr = flW + (size_t)c * 11;
#pragma unroll
    for (int k = 0; k < 11; ++k) s += xr[k] * wr[k];
    float v = siluf_(s);
    xf[t] = v; hf[t] = v; xb[t] = (__bf16)v;
}

// ---------------------------------------------------------------------------
// h2[e,k] = silu([edge_attr, dist] @ nn_W1.T + nn_b1), stored bf16 [E,128]
// ---------------------------------------------------------------------------
__global__ void edge_h2_kernel(const float* __restrict__ ea, const float* __restrict__ pos,
                               const int* __restrict__ ei, const float* __restrict__ w1,
                               const float* __restrict__ b1, __bf16* __restrict__ h2b, int E) {
    int t = blockIdx.x * blockDim.x + threadIdx.x;
    if (t >= E * W1O) return;
    int e = t >> 7, k = t & 127;
    int s = ei[e], d = ei[E + e];
    float dx = pos[s * 3 + 0] - pos[d * 3 + 0];
    float dy = pos[s * 3 + 1] - pos[d * 3 + 1];
    float dz = pos[s * 3 + 2] - pos[d * 3 + 2];
    float dist = sqrtf(dx * dx + dy * dy + dz * dz);
    const float* er = ea + (size_t)e * EDGE_DIM;
    const float* wr = w1 + (size_t)k * 5;
    float acc = b1[k] + er[0] * wr[0] + er[1] * wr[1] + er[2] * wr[2] + er[3] * wr[3] + dist * wr[4];
    h2b[(size_t)e * W1O + k] = (__bf16)siluf_(acc);
}

// ---------------------------------------------------------------------------
// degree (clipped at use-site): deg[dst] += 1
// ---------------------------------------------------------------------------
__global__ void deg_kernel(const int* __restrict__ ei, float* __restrict__ deg, int E) {
    int t = blockIdx.x * blockDim.x + threadIdx.x;
    if (t < E) atomicAdd(&deg[ei[E + t]], 1.0f);
}

// ---------------------------------------------------------------------------
// W_edge = h2 @ nn_W2.T + nn_b2   -> bf16 [E, 4096]
// WMMA bf16 16x16x32; one wave computes a 16(edge) x 64(col) tile, K=128.
// ---------------------------------------------------------------------------
__global__ __launch_bounds__(128)
void wedge_gemm_wmma(const __bf16* __restrict__ h2b, const __bf16* __restrict__ w2b,
                     const float* __restrict__ b2, __bf16* __restrict__ we, int E, int tilesE) {
    int lane = threadIdx.x & 31;
    int wave = threadIdx.x >> 5;
    int gw   = blockIdx.x * 4 + wave;
    const int JG = W2R / 64;                  // 64 col-groups
    int et = gw / JG;
    int jg = gw % JG;
    if (et >= tilesE) return;
    int j0   = jg * 64;
    int half = lane >> 4;                     // 0: lanes 0-15, 1: lanes 16-31
    int l15  = lane & 15;
    int erow = et * 16 + l15;
    if (erow >= E) erow = E - 1;              // clamp reads; stores guarded below
    const __bf16* Arow = h2b + (size_t)erow * W1O;
    int kbA = half * 16;                      // A-matrix K base per lane half
    int kbB = half * 16;                      // B-matrix K base per lane half

    v8f acc[4];
#pragma unroll
    for (int jt = 0; jt < 4; ++jt)
#pragma unroll
        for (int i = 0; i < 8; ++i) acc[jt][i] = 0.0f;

#pragma unroll
    for (int kk = 0; kk < W1O; kk += 32) {
        v16bf a;
#pragma unroll
        for (int e2 = 0; e2 < 16; ++e2) {
            int K = kbA + e2;                         // contiguous K order, matches B
            a[e2] = Arow[kk + K];
        }
#pragma unroll
        for (int jt = 0; jt < 4; ++jt) {
            int jc = j0 + jt * 16 + l15;              // B col = lane&15
            const __bf16* Brow = w2b + (size_t)jc * W1O + kk + kbB;
            v16bf b;
#pragma unroll
            for (int e2 = 0; e2 < 16; ++e2) b[e2] = Brow[e2];
            acc[jt] = __builtin_amdgcn_wmma_f32_16x16x32_bf16(
                false, a, false, b, (short)0, acc[jt], false, false);
        }
    }
#pragma unroll
    for (int jt = 0; jt < 4; ++jt) {
        int col = j0 + jt * 16 + l15;
        float bias = b2[col];
#pragma unroll
        for (int r = 0; r < 8; ++r) {
            int er = et * 16 + r + half * 8;          // C/D layout: M = r + 8*half
            if (er < E) we[(size_t)er * W2R + col] = (__bf16)(acc[jt][r] + bias);
        }
    }
}

// ---------------------------------------------------------------------------
// messages: m_e = x[src] @ W_e  (1x64 @ 64x64, bf16 in / f32 acc),
// scatter-add into agg[dst]. One wave per edge; lane owns 2 output channels.
// ---------------------------------------------------------------------------
__global__ __launch_bounds__(256)
void message_kernel(const __bf16* __restrict__ xb, const __bf16* __restrict__ we,
                    const int* __restrict__ ei, float* __restrict__ agg, int E) {
    int wave = (blockIdx.x * (blockDim.x >> 5)) + (threadIdx.x >> 5);
    if (wave >= E) return;
    int lane = threadIdx.x & 31;
    int src = ei[wave], dst = ei[E + wave];
    const __bf16* xs = xb + (size_t)src * HID;
    const __bf16* W  = we + (size_t)wave * W2R;
    float a0 = 0.0f, a1 = 0.0f;
    int o0 = lane * 2;
#pragma unroll 8
    for (int i = 0; i < HID; ++i) {
        float xi = (float)xs[i];
        const __bf16* wr = W + i * HID + o0;          // coalesced across lanes
        a0 = fmaf(xi, (float)wr[0], a0);
        a1 = fmaf(xi, (float)wr[1], a1);
    }
    atomicAdd(&agg[(size_t)dst * HID + o0], a0);
    atomicAdd(&agg[(size_t)dst * HID + o0 + 1], a1);
}

// ---------------------------------------------------------------------------
// per-node: xc = silu(x@rootW.T + rootb + agg/deg); GRU cell; x=h=new
// one 64-thread block per node
// ---------------------------------------------------------------------------
__global__ __launch_bounds__(64)
void node_update_kernel(float* __restrict__ xf, float* __restrict__ hf, __bf16* __restrict__ xb,
                        const float* __restrict__ agg, const float* __restrict__ deg,
                        const float* __restrict__ rootW, const float* __restrict__ rootb,
                        const float* __restrict__ Wih, const float* __restrict__ Whh,
                        const float* __restrict__ bih, const float* __restrict__ bhh, int N) {
    int n = blockIdx.x;
    if (n >= N) return;
    int c = threadIdx.x;
    __shared__ float xc[HID];
    __shared__ float hs[HID];
    float dv = fmaxf(deg[n], 1.0f);
    float s = rootb[c] + agg[(size_t)n * HID + c] / dv;
    const float* xr = xf + (size_t)n * HID;
    const float* wr = rootW + (size_t)c * HID;
#pragma unroll 8
    for (int k = 0; k < HID; ++k) s += xr[k] * wr[k];
    xc[c] = siluf_(s);
    hs[c] = hf[(size_t)n * HID + c];
    __syncthreads();
    float ir = bih[c], iz = bih[HID + c], in_ = bih[2 * HID + c];
    float hr = bhh[c], hz = bhh[HID + c], hn  = bhh[2 * HID + c];
    const float* wi0 = Wih + (size_t)c * HID;
    const float* wi1 = Wih + (size_t)(HID + c) * HID;
    const float* wi2 = Wih + (size_t)(2 * HID + c) * HID;
    const float* wh0 = Whh + (size_t)c * HID;
    const float* wh1 = Whh + (size_t)(HID + c) * HID;
    const float* wh2 = Whh + (size_t)(2 * HID + c) * HID;
#pragma unroll 4
    for (int k = 0; k < HID; ++k) {
        float xv = xc[k], hv = hs[k];
        ir += xv * wi0[k]; iz += xv * wi1[k]; in_ += xv * wi2[k];
        hr += hv * wh0[k]; hz += hv * wh1[k]; hn  += hv * wh2[k];
    }
    float r = sigmoidf_(ir + hr);
    float z = sigmoidf_(iz + hz);
    float nn = tanhf(in_ + r * hn);
    float hnew = (1.0f - z) * nn + z * hs[c];
    xf[(size_t)n * HID + c] = hnew;
    hf[(size_t)n * HID + c] = hnew;
    xb[(size_t)n * HID + c] = (__bf16)hnew;
}

// ---------------------------------------------------------------------------
// Set2Set LSTM cell: thread per (graph b, channel c)
// ---------------------------------------------------------------------------
__global__ void lstm_kernel(const float* __restrict__ qstar, const float* __restrict__ hlin,
                            float* __restrict__ cl, float* __restrict__ hlout,
                            const float* __restrict__ Wih, const float* __restrict__ Whh,
                            const float* __restrict__ bih, const float* __restrict__ bhh, int B) {
    int t = blockIdx.x * blockDim.x + threadIdx.x;
    if (t >= B * HID) return;
    int b = t >> 6, c = t & 63;
    float gi = bih[c] + bhh[c];
    float gf = bih[HID + c] + bhh[HID + c];
    float gg = bih[2 * HID + c] + bhh[2 * HID + c];
    float go = bih[3 * HID + c] + bhh[3 * HID + c];
    const float* q = qstar + (size_t)b * 2 * HID;
    const float* wi = Wih + (size_t)c * 2 * HID;
    const float* wf = Wih + (size_t)(HID + c) * 2 * HID;
    const float* wg = Wih + (size_t)(2 * HID + c) * 2 * HID;
    const float* wo = Wih + (size_t)(3 * HID + c) * 2 * HID;
#pragma unroll 4
    for (int k = 0; k < 2 * HID; ++k) {
        float qv = q[k];
        gi += qv * wi[k]; gf += qv * wf[k]; gg += qv * wg[k]; go += qv * wo[k];
    }
    const float* h = hlin + (size_t)b * HID;
    const float* hi = Whh + (size_t)c * HID;
    const float* hff = Whh + (size_t)(HID + c) * HID;
    const float* hg = Whh + (size_t)(2 * HID + c) * HID;
    const float* ho = Whh + (size_t)(3 * HID + c) * HID;
#pragma unroll 4
    for (int k = 0; k < HID; ++k) {
        float hv = h[k];
        gi += hv * hi[k]; gf += hv * hff[k]; gg += hv * hg[k]; go += hv * ho[k];
    }
    float cn = sigmoidf_(gf) * cl[t] + sigmoidf_(gi) * tanhf(gg);
    cl[t] = cn;
    hlout[t] = sigmoidf_(go) * tanhf(cn);
}

// monotone float<->uint encoding for atomic max
__device__ __forceinline__ unsigned fenc_(float f) {
    unsigned u = __float_as_uint(f);
    return (u & 0x80000000u) ? ~u : (u | 0x80000000u);
}
__device__ __forceinline__ float fdec_(unsigned x) {
    return __uint_as_float((x & 0x80000000u) ? (x & 0x7FFFFFFFu) : ~x);
}

// e[n] = <x[n], hl[batch[n]]>, atomic segment max (encoded)
__global__ void attn_e_kernel(const float* __restrict__ xf, const float* __restrict__ hl,
                              const int* __restrict__ batch, float* __restrict__ ebuf,
                              unsigned* __restrict__ mmax, int N) {
    int n = blockIdx.x * blockDim.x + threadIdx.x;
    if (n >= N) return;
    int b = batch[n];
    const float* xr = xf + (size_t)n * HID;
    const float* hr = hl + (size_t)b * HID;
    float s = 0.0f;
#pragma unroll 8
    for (int k = 0; k < HID; ++k) s += xr[k] * hr[k];
    ebuf[n] = s;
    atomicMax(&mmax[b], fenc_(s));
}

// a[n] = exp(e - max); asum[b] += a[n]
__global__ void attn_exp_kernel(const float* __restrict__ ebuf, const unsigned* __restrict__ mmax,
                                const int* __restrict__ batch, float* __restrict__ abuf,
                                float* __restrict__ asum, int N) {
    int n = blockIdx.x * blockDim.x + threadIdx.x;
    if (n >= N) return;
    int b = batch[n];
    float a = expf(ebuf[n] - fdec_(mmax[b]));
    abuf[n] = a;
    atomicAdd(&asum[b], a);
}

// q_star[:, :64] = hl ; q_star[:, 64:] = 0
__global__ void qinit_kernel(const float* __restrict__ hl, float* __restrict__ qstar, int B) {
    int t = blockIdx.x * blockDim.x + threadIdx.x;
    if (t >= B * 2 * HID) return;
    int b = t >> 7, j = t & 127;
    qstar[t] = (j < HID) ? hl[(size_t)b * HID + j] : 0.0f;
}

// q_star[b, 64+c] += (a[n]/asum[b]) * x[n,c]
__global__ void readout_kernel(const float* __restrict__ xf, const float* __restrict__ abuf,
                               const float* __restrict__ asum, const int* __restrict__ batch,
                               float* __restrict__ qstar, int N) {
    int t = blockIdx.x * blockDim.x + threadIdx.x;
    if (t >= N * HID) return;
    int n = t >> 6, c = t & 63;
    int b = batch[n];
    float v = (abuf[n] / asum[b]) * xf[t];
    atomicAdd(&qstar[(size_t)b * 2 * HID + HID + c], v);
}

// out[b] = silu(q_star @ W1.T + b1) @ W2.T + b2 ; 64-thread block per graph
__global__ __launch_bounds__(64)
void out_kernel(const float* __restrict__ qstar, const float* __restrict__ W1,
                const float* __restrict__ b1, const float* __restrict__ W2,
                const float* __restrict__ b2, float* __restrict__ out, int B) {
    int b = blockIdx.x;
    if (b >= B) return;
    int c = threadIdx.x;
    __shared__ float st[HID];
    const float* q = qstar + (size_t)b * 2 * HID;
    const float* w = W1 + (size_t)c * 2 * HID;
    float s = b1[c];
#pragma unroll 8
    for (int k = 0; k < 2 * HID; ++k) s += q[k] * w[k];
    st[c] = siluf_(s) * W2[c];
    __syncthreads();
    for (int str = 32; str > 0; str >>= 1) {
        if (c < str) st[c] += st[c + str];
        __syncthreads();
    }
    if (c == 0) out[b] = st[0] + b2[0];
}

// ---------------------------------------------------------------------------
// host-side launcher
// ---------------------------------------------------------------------------
static inline int cdiv_(long long a, long long b) { return (int)((a + b - 1) / b); }

extern "C" void kernel_launch(void* const* d_in, const int* in_sizes, int n_in,
                              void* d_out, int out_size, void* d_ws, size_t ws_size,
                              hipStream_t stream) {
    (void)n_in; (void)ws_size;
    const float* x_in      = (const float*)d_in[0];
    const float* edge_attr = (const float*)d_in[1];
    const float* pos       = (const float*)d_in[2];
    const int*   edge_idx  = (const int*)d_in[3];
    const int*   batch     = (const int*)d_in[4];
    const float* fl_W  = (const float*)d_in[5];
    const float* fl_b  = (const float*)d_in[6];
    const float* nn_W1 = (const float*)d_in[7];
    const float* nn_b1 = (const float*)d_in[8];
    const float* nn_W2 = (const float*)d_in[9];
    const float* nn_b2 = (const float*)d_in[10];
    const float* root_W = (const float*)d_in[11];
    const float* root_b = (const float*)d_in[12];
    const float* gru_Wih = (const float*)d_in[13];
    const float* gru_Whh = (const float*)d_in[14];
    const float* gru_bih = (const float*)d_in[15];
    const float* gru_bhh = (const float*)d_in[16];
    const float* lstm_Wih = (const float*)d_in[17];
    const float* lstm_Whh = (const float*)d_in[18];
    const float* lstm_bih = (const float*)d_in[19];
    const float* lstm_bhh = (const float*)d_in[20];
    const float* out_W1 = (const float*)d_in[21];
    const float* out_b1 = (const float*)d_in[22];
    const float* out_W2 = (const float*)d_in[23];
    const float* out_b2 = (const float*)d_in[24];

    const int N = in_sizes[0] / 11;
    const int E = in_sizes[1] / EDGE_DIM;
    const int B = out_size;

    // workspace carve-out (256B aligned)
    char* p = (char*)d_ws;
    auto take = [&](size_t bytes) -> char* {
        char* r = p;
        p += (bytes + 255) & ~(size_t)255;
        return r;
    };
    __bf16* we   = (__bf16*)take((size_t)E * W2R * sizeof(__bf16));   // 410 MB
    __bf16* h2b  = (__bf16*)take((size_t)E * W1O * sizeof(__bf16));
    __bf16* w2b  = (__bf16*)take((size_t)W2R * W1O * sizeof(__bf16));
    __bf16* xb   = (__bf16*)take((size_t)N * HID * sizeof(__bf16));
    float*  xf   = (float*)take((size_t)N * HID * sizeof(float));
    float*  hf   = (float*)take((size_t)N * HID * sizeof(float));
    float*  agg  = (float*)take((size_t)N * HID * sizeof(float));
    float*  deg  = (float*)take((size_t)N * sizeof(float));
    float*  ebuf = (float*)take((size_t)N * sizeof(float));
    float*  abuf = (float*)take((size_t)N * sizeof(float));
    float*  attn = (float*)take((size_t)2 * B * sizeof(float));  // asum | mmax(enc)
    float*  asum = attn;
    unsigned* mmax = (unsigned*)(attn + B);
    float*  qstar = (float*)take((size_t)B * 2 * HID * sizeof(float));
    float*  hlA   = (float*)take((size_t)B * HID * sizeof(float));
    float*  hlB   = (float*)take((size_t)B * HID * sizeof(float));
    float*  cl    = (float*)take((size_t)B * HID * sizeof(float));

    const int T = 256;

    // ---- pre-pass -------------------------------------------------------
    conv_w2_kernel<<<cdiv_((long long)W2R * W1O, T), T, 0, stream>>>(nn_W2, w2b, W2R * W1O);
    node_embed_kernel<<<cdiv_((long long)N * HID, T), T, 0, stream>>>(x_in, fl_W, fl_b, xf, hf, xb, N);
    edge_h2_kernel<<<cdiv_((long long)E * W1O, T), T, 0, stream>>>(edge_attr, pos, edge_idx, nn_W1, nn_b1, h2b, E);
    zero_f32_kernel<<<cdiv_(N, T), T, 0, stream>>>(deg, (size_t)N);
    deg_kernel<<<cdiv_(E, T), T, 0, stream>>>(edge_idx, deg, E);

    // ---- W_edge GEMM (WMMA bf16) ---------------------------------------
    int tilesE = cdiv_(E, 16);
    long long waves = (long long)tilesE * (W2R / 64);
    wedge_gemm_wmma<<<cdiv_(waves, 4), 128, 0, stream>>>(h2b, w2b, nn_b2, we, E, tilesE);

    // ---- 4 message-passing layers --------------------------------------
    for (int layer = 0; layer < 4; ++layer) {
        zero_f32_kernel<<<cdiv_((long long)N * HID, T), T, 0, stream>>>(agg, (size_t)N * HID);
        message_kernel<<<cdiv_(E, 8), 256, 0, stream>>>(xb, we, edge_idx, agg, E);
        node_update_kernel<<<N, 64, 0, stream>>>(xf, hf, xb, agg, deg, root_W, root_b,
                                                 gru_Wih, gru_Whh, gru_bih, gru_bhh, N);
    }

    // ---- Set2Set readout ------------------------------------------------
    zero_f32_kernel<<<cdiv_((long long)B * 2 * HID, T), T, 0, stream>>>(qstar, (size_t)B * 2 * HID);
    zero_f32_kernel<<<cdiv_((long long)B * HID, T), T, 0, stream>>>(hlA, (size_t)B * HID);
    zero_f32_kernel<<<cdiv_((long long)B * HID, T), T, 0, stream>>>(cl, (size_t)B * HID);

    float* hin = hlA;
    float* hout = hlB;
    for (int step = 0; step < 3; ++step) {
        lstm_kernel<<<cdiv_((long long)B * HID, T), T, 0, stream>>>(qstar, hin, cl, hout,
                                                                    lstm_Wih, lstm_Whh, lstm_bih, lstm_bhh, B);
        zero_f32_kernel<<<cdiv_(2 * B, T), T, 0, stream>>>(attn, (size_t)2 * B);
        attn_e_kernel<<<cdiv_(N, T), T, 0, stream>>>(xf, hout, batch, ebuf, mmax, N);
        attn_exp_kernel<<<cdiv_(N, T), T, 0, stream>>>(ebuf, mmax, batch, abuf, asum, N);
        qinit_kernel<<<cdiv_((long long)B * 2 * HID, T), T, 0, stream>>>(hout, qstar, B);
        readout_kernel<<<cdiv_((long long)N * HID, T), T, 0, stream>>>(xf, abuf, asum, batch, qstar, N);
        float* tmp = hin; hin = hout; hout = tmp;
    }

    out_kernel<<<B, 64, 0, stream>>>(qstar, out_W1, out_b1, out_W2, out_b2, (float*)d_out, B);
}
